// MixtralExpertCapacityTop2MLP_24885040513467
// MI455X (gfx1250) — matrix operand order repaired and checked by reading, not compiled, not run on hardware
//
#include <hip/hip_runtime.h>

// MoE SwiGLU MLP for MI455X (gfx1250, wave32, WMMA).
//   E=8 experts, N=B*C=2048 tokens/expert, M=2048, H=7168. fp32 in/out.
// Compute-bound (AI ~857 FLOP/B): use bf16 WMMA (v_wmma_f32_16x16x32_bf16),
// converting fp32 operands to bf16 on the fly via native v_cvt_pk_bf16_f32.
// Kernel A: hidden = silu(X*W1) * (X*W3)  -> bf16 in d_ws [E,2048,7168]
// Kernel B: out    = hidden * W2          -> f32 d_out    [E,2048,2048]

#define EXPERTS 8
#define NTOK    2048   // B*C
#define MDIM    2048
#define HDIM    7168
#define BM      128
#define BN      64
#define BK      32
#define LDA     20     // uint (bf16-pair) stride: 16 pairs + 4 pad, keeps 16B align
#define LDB     20

typedef __attribute__((ext_vector_type(16))) __bf16 v16bf;
typedef __attribute__((ext_vector_type(8)))  float  v8f;
typedef __attribute__((ext_vector_type(2)))  __bf16 bf16x2;
typedef __attribute__((ext_vector_type(2)))  float  f32x2;

union Frag {
  v16bf v;
  uint4 q[2];
};

// Packed f32x2 -> bf16x2 (RNE) -> one dword. Lowers to v_cvt_pk_bf16_f32.
__device__ __forceinline__ unsigned bfpack(float lo, float hi) {
  f32x2 f = {lo, hi};
  bf16x2 h = __builtin_convertvector(f, bf16x2);
  return __builtin_bit_cast(unsigned, h);
}

__device__ __forceinline__ unsigned short f2bf(float f) {
  return __builtin_bit_cast(unsigned short, (__bf16)f);
}

// ---------------------------------------------------------------------------
// Kernel A: gate/up GEMMs fused + SwiGLU epilogue, bf16 output.
// Block: 256 thr (8 waves, 4x2), tile BMxBN = 128x64, K-step 32.
// ---------------------------------------------------------------------------
__global__ __launch_bounds__(256)
void moe_gateup(const float* __restrict__ X,    // [E, NTOK, MDIM]
                const float* __restrict__ W1,   // [E, MDIM, HDIM]
                const float* __restrict__ W3,   // [E, MDIM, HDIM]
                unsigned short* __restrict__ Hid) // [E, NTOK, HDIM] bf16
{
  __shared__ __align__(16) unsigned As [BM * LDA]; // X tile, bf16 pairs, row-major
  __shared__ __align__(16) unsigned Bs1[BN * LDB]; // W1 tile, transposed [n][kpair]
  __shared__ __align__(16) unsigned Bs3[BN * LDB]; // W3 tile

  const int e    = blockIdx.z;
  const int n0   = blockIdx.y * BM;
  const int h0   = blockIdx.x * BN;
  const int tid  = threadIdx.x;
  const int lane = tid & 31;
  const int wid  = tid >> 5;
  const int wm   = wid & 3;   // wave row group (32 rows each)
  const int wn   = wid >> 2;  // wave col group (32 cols each)
  const int lx   = lane & 15;
  const int hi   = lane >> 4;

  const float* Xe  = X  + (size_t)e * NTOK * MDIM;
  const float* W1e = W1 + (size_t)e * MDIM * HDIM;
  const float* W3e = W3 + (size_t)e * MDIM * HDIM;

  // per-thread staging slots
  const int kp = tid >> 4;          // 0..15: K pair-row in W tile (k = 2*kp)
  const int nb = (tid & 15) << 2;   // 0,4,...,60: base col in W tile

  float4 xr[4], w1lo, w1hi, w3lo, w3hi;

  auto loadGlb = [&](int k0) {
#pragma unroll
    for (int i = 0; i < 4; ++i) {
      int p = tid + (i << 8);       // 0..1023 float4 slots of X tile
      int r = p >> 3, c4 = p & 7;
      xr[i] = *(const float4*)(Xe + (size_t)(n0 + r) * MDIM + k0 + (c4 << 2));
    }
    const float* w1p = W1e + (size_t)(k0 + 2 * kp) * HDIM + h0 + nb;
    const float* w3p = W3e + (size_t)(k0 + 2 * kp) * HDIM + h0 + nb;
    w1lo = *(const float4*)(w1p);
    w1hi = *(const float4*)(w1p + HDIM);
    w3lo = *(const float4*)(w3p);
    w3hi = *(const float4*)(w3p + HDIM);
  };

  auto storeLds = [&]() {
#pragma unroll
    for (int i = 0; i < 4; ++i) {
      int p = tid + (i << 8);
      int r = p >> 3, c4 = p & 7;
      As[r * LDA + (c4 << 1)    ] = bfpack(xr[i].x, xr[i].y);
      As[r * LDA + (c4 << 1) + 1] = bfpack(xr[i].z, xr[i].w);
    }
    Bs1[(nb + 0) * LDB + kp] = bfpack(w1lo.x, w1hi.x);
    Bs1[(nb + 1) * LDB + kp] = bfpack(w1lo.y, w1hi.y);
    Bs1[(nb + 2) * LDB + kp] = bfpack(w1lo.z, w1hi.z);
    Bs1[(nb + 3) * LDB + kp] = bfpack(w1lo.w, w1hi.w);
    Bs3[(nb + 0) * LDB + kp] = bfpack(w3lo.x, w3hi.x);
    Bs3[(nb + 1) * LDB + kp] = bfpack(w3lo.y, w3hi.y);
    Bs3[(nb + 2) * LDB + kp] = bfpack(w3lo.z, w3hi.z);
    Bs3[(nb + 3) * LDB + kp] = bfpack(w3lo.w, w3hi.w);
  };

  v8f accG[2][2], accU[2][2];
#pragma unroll
  for (int i = 0; i < 2; ++i)
#pragma unroll
    for (int j = 0; j < 2; ++j) { accG[i][j] = {}; accU[i][j] = {}; }

  loadGlb(0);

  for (int k0 = 0; k0 < MDIM; k0 += BK) {
    __syncthreads();
    storeLds();
    __syncthreads();

    if (k0 + BK < MDIM) {
      loadGlb(k0 + BK);                                   // overlap with math
      if (k0 + 2 * BK < MDIM) {                           // warm L2 two steps out
        __builtin_prefetch(W1e + (size_t)(k0 + 2 * BK) * HDIM + h0, 0, 1);
        __builtin_prefetch(W3e + (size_t)(k0 + 2 * BK) * HDIM + h0, 0, 1);
      }
    }

    Frag a[2], b1[2], b3[2];
#pragma unroll
    for (int i = 0; i < 2; ++i) {
      int row = wm * 32 + i * 16 + lx;
      a[i].q[0] = *(const uint4*)&As[row * LDA + (hi << 2)];       // K 0-7 / 8-15
      a[i].q[1] = *(const uint4*)&As[row * LDA + 8 + (hi << 2)];   // K 16-23 / 24-31
    }
#pragma unroll
    for (int j = 0; j < 2; ++j) {
      int col = wn * 32 + j * 16 + lx;
      b1[j].q[0] = *(const uint4*)&Bs1[col * LDB + (hi << 2)];     // K 0-7 / 8-15
      b1[j].q[1] = *(const uint4*)&Bs1[col * LDB + 8 + (hi << 2)];
      b3[j].q[0] = *(const uint4*)&Bs3[col * LDB + (hi << 2)];
      b3[j].q[1] = *(const uint4*)&Bs3[col * LDB + 8 + (hi << 2)];
    }

#pragma unroll
    for (int i = 0; i < 2; ++i)
#pragma unroll
      for (int j = 0; j < 2; ++j) {
        accG[i][j] = __builtin_amdgcn_wmma_f32_16x16x32_bf16(
            false, a[i].v, false, b1[j].v, (short)0, accG[i][j], false, false);
        accU[i][j] = __builtin_amdgcn_wmma_f32_16x16x32_bf16(
            false, a[i].v, false, b3[j].v, (short)0, accU[i][j], false, false);
      }
  }

  // SwiGLU epilogue: hidden = silu(gate) * up, store bf16.
  // C/D layout: VGPR v, lanes 0-15 -> M=v, lanes 16-31 -> M=v+8; N=lane&15.
  const size_t hidBase = (size_t)e * NTOK * HDIM;
#pragma unroll
  for (int i = 0; i < 2; ++i)
#pragma unroll
    for (int j = 0; j < 2; ++j) {
      int col = h0 + wn * 32 + j * 16 + lx;
#pragma unroll
      for (int v = 0; v < 8; ++v) {
        int row = n0 + wm * 32 + i * 16 + v + (hi << 3);
        float g = accG[i][j][v];
        float u = accU[i][j][v];
        float s = g / (1.0f + __expf(-g));   // silu
        Hid[hidBase + (size_t)row * HDIM + col] = f2bf(s * u);
      }
    }
}

// ---------------------------------------------------------------------------
// Kernel B: out = hidden(bf16) * W2(f32->bf16), f32 output.
// ---------------------------------------------------------------------------
__global__ __launch_bounds__(256)
void moe_down(const unsigned* __restrict__ Hid, // [E, NTOK, HDIM/2] bf16 pairs
              const float* __restrict__ W2,     // [E, HDIM, MDIM]
              float* __restrict__ Out)          // [E, NTOK, MDIM]
{
  __shared__ __align__(16) unsigned As[BM * LDA];
  __shared__ __align__(16) unsigned Bs[BN * LDB];

  const int e    = blockIdx.z;
  const int n0   = blockIdx.y * BM;
  const int m0   = blockIdx.x * BN;
  const int tid  = threadIdx.x;
  const int lane = tid & 31;
  const int wid  = tid >> 5;
  const int wm   = wid & 3;
  const int wn   = wid >> 2;
  const int lx   = lane & 15;
  const int hi   = lane >> 4;

  const unsigned* He = Hid + (size_t)e * NTOK * (HDIM / 2);
  const float*   W2e = W2  + (size_t)e * HDIM * MDIM;

  const int kp = tid >> 4;
  const int nb = (tid & 15) << 2;

  uint4  hr[2];
  float4 w2lo, w2hi;

  auto loadGlb = [&](int k0) {
#pragma unroll
    for (int i = 0; i < 2; ++i) {
      int p = tid + (i << 8);          // 512 uint4 slots (128 rows x 4)
      int r = p >> 2, c = p & 3;
      hr[i] = *(const uint4*)(He + (size_t)(n0 + r) * (HDIM / 2) + (k0 >> 1) + (c << 2));
    }
    const float* wp = W2e + (size_t)(k0 + 2 * kp) * MDIM + m0 + nb;
    w2lo = *(const float4*)(wp);
    w2hi = *(const float4*)(wp + MDIM);
  };

  auto storeLds = [&]() {
#pragma unroll
    for (int i = 0; i < 2; ++i) {
      int p = tid + (i << 8);
      int r = p >> 2, c = p & 3;
      *(uint4*)&As[r * LDA + (c << 2)] = hr[i];       // already bf16 pairs
    }
    Bs[(nb + 0) * LDB + kp] = bfpack(w2lo.x, w2hi.x);
    Bs[(nb + 1) * LDB + kp] = bfpack(w2lo.y, w2hi.y);
    Bs[(nb + 2) * LDB + kp] = bfpack(w2lo.z, w2hi.z);
    Bs[(nb + 3) * LDB + kp] = bfpack(w2lo.w, w2hi.w);
  };

  v8f acc[2][2];
#pragma unroll
  for (int i = 0; i < 2; ++i)
#pragma unroll
    for (int j = 0; j < 2; ++j) acc[i][j] = {};

  loadGlb(0);

  for (int k0 = 0; k0 < HDIM; k0 += BK) {
    __syncthreads();
    storeLds();
    __syncthreads();

    if (k0 + BK < HDIM) {
      loadGlb(k0 + BK);
      if (k0 + 2 * BK < HDIM)
        __builtin_prefetch(W2e + (size_t)(k0 + 2 * BK) * MDIM + m0, 0, 1);
    }

    Frag a[2], b[2];
#pragma unroll
    for (int i = 0; i < 2; ++i) {
      int row = wm * 32 + i * 16 + lx;
      a[i].q[0] = *(const uint4*)&As[row * LDA + (hi << 2)];
      a[i].q[1] = *(const uint4*)&As[row * LDA + 8 + (hi << 2)];
    }
#pragma unroll
    for (int j = 0; j < 2; ++j) {
      int col = wn * 32 + j * 16 + lx;
      b[j].q[0] = *(const uint4*)&Bs[col * LDB + (hi << 2)];
      b[j].q[1] = *(const uint4*)&Bs[col * LDB + 8 + (hi << 2)];
    }

#pragma unroll
    for (int i = 0; i < 2; ++i)
#pragma unroll
      for (int j = 0; j < 2; ++j)
        acc[i][j] = __builtin_amdgcn_wmma_f32_16x16x32_bf16(
            false, a[i].v, false, b[j].v, (short)0, acc[i][j], false, false);
  }

  const size_t outBase = (size_t)e * NTOK * MDIM;
#pragma unroll
  for (int i = 0; i < 2; ++i)
#pragma unroll
    for (int j = 0; j < 2; ++j) {
      int col = m0 + wn * 32 + j * 16 + lx;
#pragma unroll
      for (int v = 0; v < 8; ++v) {
        int row = n0 + wm * 32 + i * 16 + v + (hi << 3);
        Out[outBase + (size_t)row * MDIM + col] = acc[i][j][v];
      }
    }
}

// ---------------------------------------------------------------------------
extern "C" void kernel_launch(void* const* d_in, const int* in_sizes, int n_in,
                              void* d_out, int out_size, void* d_ws, size_t ws_size,
                              hipStream_t stream) {
  const float* X  = (const float*)d_in[0];  // dispatch_input [E,B,C,M]
  const float* W1 = (const float*)d_in[1];  // [E,M,H]
  const float* W2 = (const float*)d_in[2];  // [E,H,M]
  const float* W3 = (const float*)d_in[3];  // [E,M,H]

  unsigned short* Hid = (unsigned short*)d_ws;  // [E,NTOK,HDIM] bf16, 235 MB

  dim3 blk(256, 1, 1);
  dim3 gA(HDIM / BN, NTOK / BM, EXPERTS);   // 112 x 16 x 8 (expert slowest -> L2 reuse)
  moe_gateup<<<gA, blk, 0, stream>>>(X, W1, W3, Hid);

  dim3 gB(MDIM / BN, NTOK / BM, EXPERTS);   // 32 x 16 x 8
  moe_down<<<gB, blk, 0, stream>>>((const unsigned*)Hid, W2, (float*)d_out);
}
